// GlobalLocalSparseAttention_57698590655268
// MI455X (gfx1250) — compile-verified, matrix operand order checked by reading, not run on hardware
//
#include <hip/hip_runtime.h>
#include <hip/hip_bf16.h>

typedef __attribute__((ext_vector_type(16))) _Float16 v16h;
typedef __attribute__((ext_vector_type(8)))  float    v8f;

#define DEV __device__ __forceinline__

constexpr int S_    = 2048;
constexpr int D_    = 1024;
constexpr int H_    = 16;
constexpr int HKV_  = 4;
constexpr int DH_   = 64;
constexpr int G_    = 4;
constexpr int BS_   = 32;
constexpr int NSEL_ = 8;
constexpr int MMEM_ = 4;
constexpr int WIN_  = 64;
constexpr int W_    = 64;                 // S / BS
constexpr int CHID_ = 2048;               // BS*DH
constexpr int CKP_  = 96;                 // padded (MMEM + W = 68) -> 96
constexpr int SELK_ = (NSEL_ + 1) * BS_;  // 288
constexpr float SCALE_ = 0.125f;          // 64^-0.5
constexpr float NEG_   = -1e30f;

DEV v8f zero8() { v8f z = {0.f,0.f,0.f,0.f,0.f,0.f,0.f,0.f}; return z; }

DEV v8f wmma32(v16h a, v16h b, v8f c) {
  return __builtin_amdgcn_wmma_f32_16x16x32_f16(false, a, false, b, (short)0, c, false, false);
}

// CDNA5 async copy global -> LDS (ASYNCcnt tracked).  VDST = LDS byte offset
// VGPR, VADDR = 64-bit global address (ISA 15.18.3 op 98).
DEV void async_g2l_b128(unsigned lds_off, const void* gaddr) {
  asm volatile("global_load_async_to_lds_b128 %0, %1, off"
               :: "v"(lds_off), "v"(gaddr) : "memory");
}
DEV void wait_async_le2() { asm volatile("s_wait_asynccnt 0x2" ::: "memory"); }
DEV void wait_async_le0() { asm volatile("s_wait_asynccnt 0x0" ::: "memory"); }

// ---------------------------------------------------------------- utility ---
__global__ void k_zero(float* p, int n) {
  int i = blockIdx.x * 256 + threadIdx.x;
  if (i < n) p[i] = 0.f;
}

__global__ void k_rmsnorm(const float* __restrict__ x, const float* __restrict__ w,
                          float* __restrict__ h32, _Float16* __restrict__ h16) {
  int s = blockIdx.x, t = threadIdx.x;
  const float* row = x + (size_t)s * D_;
  float ss = 0.f;
  for (int d = t; d < D_; d += 256) { float v = row[d]; ss += v * v; }
  __shared__ float red[256];
  red[t] = ss; __syncthreads();
  for (int k = 128; k > 0; k >>= 1) { if (t < k) red[t] += red[t + k]; __syncthreads(); }
  float inv = rsqrtf(red[0] / (float)D_ + 1e-6f);
  for (int d = t; d < D_; d += 256) {
    float v = row[d] * inv * w[d];
    h32[(size_t)s * D_ + d] = v;
    h16[(size_t)s * D_ + d] = (_Float16)v;
  }
}

// dst[n][k] (f16) = src[k][n] (f32)
__global__ void k_transpose_cvt(const float* __restrict__ src, _Float16* __restrict__ dst,
                                int K, int N) {
  int i = blockIdx.x * 256 + threadIdx.x;
  if (i >= K * N) return;
  int k = i / N, n = i % N;
  dst[(size_t)n * K + k] = (_Float16)src[i];
}

// ------------------------------------------------------------------- GEMM ---
// C[M][N] = A[M][K] * Bt[N][K]^T.  Block = 128 threads (4 waves), block tile
// 128(M) x 64(N).  Wave w owns rows [tm, tm+32) as two 16-row tiles; all
// waves share one 64x32 B tile staged in LDS via double-buffered
// GLOBAL_LOAD_ASYNC_TO_LDS_B128.  8 WMMA per wave per k-step, branch-free
// inner loop (N-edge handled by clamped staging + epilogue guard).
// Requirements (call sites satisfy): M % 128 == 0, N % 16 == 0, K % 32 == 0.
__global__ void k_gemm(const _Float16* __restrict__ A, int lda,
                       const _Float16* __restrict__ Bt, int ldb,
                       float* __restrict__ C32, _Float16* __restrict__ C16, int ldc,
                       const float* __restrict__ bias, int relu,
                       int M, int N, int K) {
  __shared__ __align__(64) _Float16 bstage[2][64][32];
  int tid = threadIdx.x;
  int lane = tid & 31, wv = tid >> 5;
  int tm  = blockIdx.y * 128 + wv * 32;
  int tn0 = blockIdx.x * 64;
  int l = lane & 15, hi = lane >> 4;
  unsigned lbase = (unsigned)(size_t)(&bstage[0][0][0]);

  const _Float16* ar0 = A + (size_t)(tm + l) * lda + 16 * hi;
  const _Float16* ar1 = ar0 + (size_t)16 * lda;

  // stage B rows [tn0, tn0+64) x 32 K-halves for k-step `kt` into buffer `buf`
  auto issue = [&](int kt, int buf) {
    int k = kt * 32;
    #pragma unroll
    for (int h = 0; h < 2; ++h) {
      int c = tid + h * 128;            // 256 16B chunks: 64 rows x 4 chunks
      int row = c >> 2, ch = c & 3;
      int grow = tn0 + row;
      if (grow > N - 1) grow = N - 1;   // clamp (only hits when N < tn0+64)
      const void* g = (const void*)(Bt + (size_t)grow * ldb + k + ch * 8);
      unsigned loff = lbase + (unsigned)(((buf * 64 + row) * 32 + ch * 8) * 2);
      async_g2l_b128(loff, g);
    }
  };

  v8f acc[2][4];
  #pragma unroll
  for (int i = 0; i < 2; i++)
    #pragma unroll
    for (int t = 0; t < 4; t++) acc[i][t] = zero8();

  auto compute = [&](int kt) {
    int cur = kt & 1;
    v16h a0 = *(const v16h*)(ar0 + kt * 32);
    v16h a1 = *(const v16h*)(ar1 + kt * 32);
    __builtin_prefetch(ar0 + kt * 32 + 256, 0, 1);
    #pragma unroll
    for (int t = 0; t < 4; t++) {
      v16h b = *(const v16h*)(&bstage[cur][t * 16 + l][16 * hi]);
      acc[0][t] = wmma32(a0, b, acc[0][t]);
      acc[1][t] = wmma32(a1, b, acc[1][t]);
    }
  };

  int KT = K >> 5;
  issue(0, 0);
  for (int kt = 0; kt < KT - 1; ++kt) {
    issue(kt + 1, (kt & 1) ^ 1);
    wait_async_le2();
    __syncthreads();                    // B tile visible to all 4 waves
    compute(kt);
    __syncthreads();                    // done reading before buffer reuse
  }
  wait_async_le0();
  __syncthreads();
  compute(KT - 1);

  #pragma unroll
  for (int i = 0; i < 2; i++)
    #pragma unroll
    for (int t = 0; t < 4; t++) {
      int tn = tn0 + t * 16;
      if (tn >= N) continue;
      #pragma unroll
      for (int r = 0; r < 8; r++) {
        int m = tm + i * 16 + r + 8 * hi, n = tn + l;
        float v = acc[i][t][r];
        if (bias) v += bias[n];
        if (relu) v = fmaxf(v, 0.f);
        size_t o = (size_t)m * ldc + n;
        if (C32) C32[o] = v;
        if (C16) C16[o] = (_Float16)v;
      }
    }
}

// ------------------------------------------------------- rotary + packing ---
__global__ void k_rope_pack(const float* __restrict__ q32, const float* __restrict__ k32,
                            const float* __restrict__ v32,
                            _Float16* __restrict__ q16, _Float16* __restrict__ qr16,
                            _Float16* __restrict__ kr16, _Float16* __restrict__ v16o) {
  int s = blockIdx.x, t = threadIdx.x;          // 512 threads
  int p = t & 31;
  float freq = __expf(-(float)(2 * p) / (float)DH_ * 9.210340371976184f); // ln(1e4)
  float ang = (float)s * freq;
  float c = __cosf(ang), sn = __sinf(ang);
  {
    int hh = t >> 5;
    size_t b = (size_t)s * D_ + hh * 64 + 2 * p;
    float x0 = q32[b], x1 = q32[b + 1];
    q16[b] = (_Float16)x0; q16[b + 1] = (_Float16)x1;
    qr16[b]     = (_Float16)(x0 * c - x1 * sn);
    qr16[b + 1] = (_Float16)(x0 * sn + x1 * c);
  }
  if (t < 128) {
    int h2 = t >> 5;
    size_t kb = (size_t)s * 256 + h2 * 64 + 2 * p;
    float y0 = k32[kb], y1 = k32[kb + 1];
    kr16[kb]     = (_Float16)(y0 * c - y1 * sn);
    kr16[kb + 1] = (_Float16)(y0 * sn + y1 * c);
  }
  if (t < 256) {
    size_t vb = (size_t)s * 256 + t;
    v16o[vb] = (_Float16)v32[vb];
  }
}

// flat[(hkv*W + w)][j*64+d] = kv[(w*32+j)][hkv*64+d] + pos[hkv][j][d]
__global__ void k_compress_in(const float* __restrict__ kv32, const float* __restrict__ pos,
                              _Float16* __restrict__ flat) {
  int i = blockIdx.x * 256 + threadIdx.x;       // HKV*W*BS*DH = 524288
  if (i >= HKV_ * W_ * BS_ * DH_) return;
  int d = i & 63, j = (i >> 6) & 31, w = (i >> 11) & 63, hkv = i >> 17;
  float v = kv32[(size_t)(w * BS_ + j) * 256 + hkv * 64 + d] +
            pos[(size_t)(hkv * BS_ + j) * 64 + d];
  flat[(size_t)(hkv * W_ + w) * (BS_ * DH_) + j * 64 + d] = (_Float16)v;
}

// build ck (row major, padded to 96 rows) and/or cv^T ([d][j] padded)
__global__ void k_pack_c(const float* __restrict__ cblk, const float* __restrict__ memkv,
                         int memsel, _Float16* __restrict__ ckRM, _Float16* __restrict__ cvT) {
  int i = blockIdx.x * 256 + threadIdx.x;       // HKV*CKP*DH = 24576
  if (i >= HKV_ * CKP_ * DH_) return;
  int d = i & 63, j = (i >> 6) % CKP_, hkv = i / (CKP_ * DH_);
  float v;
  if (j < MMEM_)               v = memkv[(size_t)((memsel * HKV_ + hkv) * MMEM_ + j) * DH_ + d];
  else if (j < MMEM_ + W_)     v = cblk[(size_t)(hkv * W_ + (j - MMEM_)) * DH_ + d];
  else                         v = 0.f;
  if (ckRM) ckRM[(size_t)(hkv * CKP_ + j) * DH_ + d] = (_Float16)v;
  if (cvT)  cvT[(size_t)(hkv * DH_ + d) * CKP_ + j] = (_Float16)v;
}

// ---------------------------------------------- branch 1: compressed attn ---
__global__ void k_compressed_attn(const _Float16* __restrict__ q16,
                                  const _Float16* __restrict__ ck,
                                  const _Float16* __restrict__ cvT,
                                  float* __restrict__ imp, float* __restrict__ cout) {
  int hkv = blockIdx.z, g = blockIdx.y, s0 = blockIdx.x * 16;
  int head = hkv * G_ + g;
  int lane = threadIdx.x & 31, l = lane & 15, hi = lane >> 4;
  __shared__ float    simL[16][CKP_];
  __shared__ _Float16 attnL[16][CKP_];
  const _Float16* ar = q16 + (size_t)(s0 + l) * D_ + head * 64 + 16 * hi;

  v8f acc[6];
  #pragma unroll
  for (int t = 0; t < 6; t++) acc[t] = zero8();
  #pragma unroll
  for (int t = 0; t < 6; t++) {
    const _Float16* br = ck + (size_t)(hkv * CKP_ + t * 16 + l) * 64 + 16 * hi;
    #pragma unroll
    for (int k = 0; k < 64; k += 32)
      acc[t] = wmma32(*(const v16h*)(ar + k), *(const v16h*)(br + k), acc[t]);
  }
  #pragma unroll
  for (int t = 0; t < 6; t++) {
    int j = t * 16 + l;
    #pragma unroll
    for (int r = 0; r < 8; r++) {
      int m = r + 8 * hi, s = s0 + m;
      float v = acc[t][r] * SCALE_;
      bool vis = (j < MMEM_) || (j < MMEM_ + W_ && s >= (j - MMEM_ + 1) * BS_ - 1);
      if (!vis) v = NEG_;
      simL[m][j] = v;
      if (j >= MMEM_ && j < MMEM_ + W_)
        atomicAdd(&imp[((size_t)hkv * S_ + s) * W_ + (j - MMEM_)], v * 0.25f);
    }
  }
  __syncthreads();
  if (threadIdx.x < 16) {
    int m = threadIdx.x;
    float mx = -3.4e38f;
    for (int j = 0; j < MMEM_ + W_; j++) mx = fmaxf(mx, simL[m][j]);
    float sum = 0.f;
    for (int j = 0; j < MMEM_ + W_; j++) { float e = __expf(simL[m][j] - mx); simL[m][j] = e; sum += e; }
    float inv = 1.f / sum;
    for (int j = 0; j < CKP_; j++)
      attnL[m][j] = (_Float16)((j < MMEM_ + W_) ? simL[m][j] * inv : 0.f);
  }
  __syncthreads();
  v8f out[4];
  #pragma unroll
  for (int t = 0; t < 4; t++) out[t] = zero8();
  #pragma unroll
  for (int t = 0; t < 4; t++) {
    const _Float16* br = cvT + (size_t)(hkv * DH_ + t * 16 + l) * CKP_;
    #pragma unroll
    for (int k = 0; k < CKP_; k += 32) {
      v16h a = *(const v16h*)(&attnL[l][16 * hi + k]);
      v16h b = *(const v16h*)(br + 16 * hi + k);
      out[t] = wmma32(a, b, out[t]);
    }
  }
  #pragma unroll
  for (int t = 0; t < 4; t++)
    #pragma unroll
    for (int r = 0; r < 8; r++)
      cout[(size_t)(s0 + r + 8 * hi) * D_ + head * 64 + t * 16 + l] = out[t][r];
}

// --------------------------------------------------------- top-k selection ---
__global__ void k_topk(const float* __restrict__ imp, int* __restrict__ idx_all,
                       int* __restrict__ dupm) {
  int s = blockIdx.x * 256 + threadIdx.x;
  int hkv = blockIdx.y;
  if (s >= S_) return;
  const float* row = imp + ((size_t)hkv * S_ + s) * W_;
  float v[W_];
  for (int w = 0; w < W_; w++) v[w] = row[w];
  unsigned long long used = 0;
  int own = s / BS_;
  int* out = idx_all + ((size_t)hkv * S_ + s) * (NSEL_ + 1);
  int dup = 0;
  for (int t = 0; t < NSEL_; t++) {
    float best = -3.4e38f; int bi = 0;
    for (int w = 0; w < W_; w++)
      if (((used >> w) & 1ull) == 0 && v[w] > best) { best = v[w]; bi = w; }
    used |= 1ull << bi;
    out[t] = bi;
    if (bi == own) dup |= 1 << t;
  }
  out[NSEL_] = own;
  dupm[(size_t)hkv * S_ + s] = dup;
}

// ---------------------------------------------- branch 2: fine (selected) ---
__global__ void k_fine_attn(const _Float16* __restrict__ qr, const _Float16* __restrict__ kr,
                            const _Float16* __restrict__ v16, const int* __restrict__ idx_all,
                            const int* __restrict__ dupm, float* __restrict__ fout) {
  int s = blockIdx.x, hkv = blockIdx.y;
  int lane = threadIdx.x & 31, l = lane & 15, hi = lane >> 4;
  __shared__ float    simL[16][SELK_];
  __shared__ _Float16 attnL[16][SELK_];
  __shared__ _Float16 vTL[64][96];
  const int* idxp = idx_all + ((size_t)hkv * S_ + s) * (NSEL_ + 1);
  int dup = dupm[(size_t)hkv * S_ + s];
  int idxv[NSEL_ + 1];
  #pragma unroll
  for (int t = 0; t <= NSEL_; t++) idxv[t] = idxp[t];
  int head = hkv * G_ + (l & 3);                // pad 4 real queries to 16 rows
  const _Float16* ar = qr + (size_t)s * D_ + head * 64 + 16 * hi;

  #pragma unroll
  for (int c = 0; c < 2; c++) {                 // 18 key tiles in 2 chunks of 9
    v8f acc[9];
    #pragma unroll
    for (int t = 0; t < 9; t++) acc[t] = zero8();
    #pragma unroll
    for (int t = 0; t < 9; t++) {
      int jt = c * 9 + t, sel = jt >> 1;
      int key = idxv[sel] * BS_ + (jt & 1) * 16 + l;
      const _Float16* br = kr + (size_t)key * 256 + hkv * 64 + 16 * hi;
      #pragma unroll
      for (int k = 0; k < 64; k += 32)
        acc[t] = wmma32(*(const v16h*)(ar + k), *(const v16h*)(br + k), acc[t]);
    }
    #pragma unroll
    for (int t = 0; t < 9; t++) {
      int jt = c * 9 + t, sel = jt >> 1;
      int kpos = idxv[sel] * BS_ + (jt & 1) * 16 + l;
      bool blocked = (sel < NSEL_) && ((dup >> sel) & 1);
      #pragma unroll
      for (int r = 0; r < 8; r++) {
        int m = r + 8 * hi;
        float vv = acc[t][r] * SCALE_;
        if (kpos > s || blocked) vv = NEG_;
        simL[m][jt * 16 + l] = vv;
      }
    }
  }
  __syncthreads();
  if (threadIdx.x < 4) {
    int m = threadIdx.x;
    float mx = -3.4e38f;
    for (int j = 0; j < SELK_; j++) mx = fmaxf(mx, simL[m][j]);
    float sum = 0.f;
    for (int j = 0; j < SELK_; j++) { float e = __expf(simL[m][j] - mx); simL[m][j] = e; sum += e; }
    float inv = 1.f / sum;
    for (int j = 0; j < SELK_; j++) attnL[m][j] = (_Float16)(simL[m][j] * inv);
  } else if (threadIdx.x < 16) {
    int m = threadIdx.x;
    for (int j = 0; j < SELK_; j++) attnL[m][j] = (_Float16)0.f;
  }
  __syncthreads();
  v8f out[4];
  #pragma unroll
  for (int t = 0; t < 4; t++) out[t] = zero8();
  for (int c = 0; c < 3; c++) {                 // attn @ V in key chunks of 96
    for (int j = lane; j < 96; j += 32) {       // stage V^T for this chunk
      int jj = c * 96 + j, sel = jj / 32;
      int key = idxv[sel] * BS_ + (jj & 31);
      const _Float16* vr = v16 + (size_t)key * 256 + hkv * 64;
      v16h t0 = *(const v16h*)(vr),      t1 = *(const v16h*)(vr + 16);
      v16h t2 = *(const v16h*)(vr + 32), t3 = *(const v16h*)(vr + 48);
      #pragma unroll
      for (int d = 0; d < 16; d++) {
        vTL[d][j] = t0[d]; vTL[d + 16][j] = t1[d];
        vTL[d + 32][j] = t2[d]; vTL[d + 48][j] = t3[d];
      }
    }
    __syncthreads();
    #pragma unroll
    for (int t = 0; t < 4; t++)
      #pragma unroll
      for (int k = 0; k < 96; k += 32) {
        v16h a = *(const v16h*)(&attnL[l][c * 96 + 16 * hi + k]);
        v16h b = *(const v16h*)(&vTL[t * 16 + l][16 * hi + k]);
        out[t] = wmma32(a, b, out[t]);
      }
    __syncthreads();
  }
  if (hi == 0)
    #pragma unroll
    for (int t = 0; t < 4; t++)
      #pragma unroll
      for (int r = 0; r < 4; r++)
        fout[(size_t)s * D_ + (hkv * G_ + r) * 64 + t * 16 + l] = out[t][r];
}

// ---------------------------------------------- branch 3: sliding window ----
__global__ void k_window_attn(const _Float16* __restrict__ qr, const _Float16* __restrict__ kr,
                              const _Float16* __restrict__ v16, float* __restrict__ sout) {
  int hkv = blockIdx.z, g = blockIdx.y, s0 = blockIdx.x * 16;
  int head = hkv * G_ + g;
  int nb = s0 >> 6;
  int lane = threadIdx.x & 31, l = lane & 15, hi = lane >> 4;
  __shared__ float    simL[16][2 * WIN_];
  __shared__ _Float16 attnL[16][2 * WIN_];
  __shared__ _Float16 vTL[64][2 * WIN_];
  const _Float16* ar = qr + (size_t)(s0 + l) * D_ + head * 64 + 16 * hi;
  int jbase = nb * WIN_ - WIN_;

  v8f acc[8];
  #pragma unroll
  for (int t = 0; t < 8; t++) acc[t] = zero8();
  #pragma unroll
  for (int t = 0; t < 8; t++) {
    int jg = jbase + t * 16 + l;
    int jgc = jg < 0 ? 0 : jg;
    const _Float16* br = kr + (size_t)jgc * 256 + hkv * 64 + 16 * hi;
    #pragma unroll
    for (int k = 0; k < 64; k += 32)
      acc[t] = wmma32(*(const v16h*)(ar + k), *(const v16h*)(br + k), acc[t]);
  }
  #pragma unroll
  for (int t = 0; t < 8; t++) {
    int kj = t * 16 + l, jg = jbase + kj;
    #pragma unroll
    for (int r = 0; r < 8; r++) {
      int m = r + 8 * hi, qg = s0 + m;
      float vv = acc[t][r] * SCALE_;
      int diff = qg - jg;
      if (jg < 0 || diff < 0 || diff > WIN_) vv = NEG_;
      simL[m][kj] = vv;
    }
  }
  for (int j = lane; j < 2 * WIN_; j += 32) {   // stage V^T
    int jg = jbase + j, jgc = jg < 0 ? 0 : jg;
    const _Float16* vr = v16 + (size_t)jgc * 256 + hkv * 64;
    v16h t0 = *(const v16h*)(vr),      t1 = *(const v16h*)(vr + 16);
    v16h t2 = *(const v16h*)(vr + 32), t3 = *(const v16h*)(vr + 48);
    #pragma unroll
    for (int d = 0; d < 16; d++) {
      vTL[d][j] = t0[d]; vTL[d + 16][j] = t1[d];
      vTL[d + 32][j] = t2[d]; vTL[d + 48][j] = t3[d];
    }
  }
  __syncthreads();
  if (threadIdx.x < 16) {
    int m = threadIdx.x;
    float mx = -3.4e38f;
    for (int j = 0; j < 2 * WIN_; j++) mx = fmaxf(mx, simL[m][j]);
    float sum = 0.f;
    for (int j = 0; j < 2 * WIN_; j++) { float e = __expf(simL[m][j] - mx); simL[m][j] = e; sum += e; }
    float inv = 1.f / sum;
    for (int j = 0; j < 2 * WIN_; j++) attnL[m][j] = (_Float16)(simL[m][j] * inv);
  }
  __syncthreads();
  v8f out[4];
  #pragma unroll
  for (int t = 0; t < 4; t++) out[t] = zero8();
  #pragma unroll
  for (int t = 0; t < 4; t++)
    #pragma unroll
    for (int k = 0; k < 2 * WIN_; k += 32) {
      v16h a = *(const v16h*)(&attnL[l][16 * hi + k]);
      v16h b = *(const v16h*)(&vTL[t * 16 + l][16 * hi + k]);
      out[t] = wmma32(a, b, out[t]);
    }
  #pragma unroll
  for (int t = 0; t < 4; t++)
    #pragma unroll
    for (int r = 0; r < 8; r++)
      sout[(size_t)(s0 + r + 8 * hi) * D_ + head * 64 + t * 16 + l] = out[t][r];
}

// --------------------------------------------------------- gated combine ----
__global__ void k_combine(const float* __restrict__ cout, const float* __restrict__ fout,
                          const float* __restrict__ sout, const float* __restrict__ gate_lin,
                          _Float16* __restrict__ ocat) {
  int i = blockIdx.x * 256 + threadIdx.x;
  if (i >= S_ * D_) return;
  int s = i >> 10, hd = i & 1023, hh = hd >> 6;
  const float* gl = gate_lin + (size_t)s * 48 + hh * 3;
  float g0 = 1.f / (1.f + __expf(-gl[0]));
  float g1 = 1.f / (1.f + __expf(-gl[1]));
  float g2 = 1.f / (1.f + __expf(-gl[2]));
  ocat[i] = (_Float16)(cout[i] * g0 + fout[i] * g1 + sout[i] * g2);
}

// ------------------------------------------------------------------ launch ---
extern "C" void kernel_launch(void* const* d_in, const int* in_sizes, int n_in,
                              void* d_out, int out_size, void* d_ws, size_t ws_size,
                              hipStream_t stream) {
  (void)in_sizes; (void)n_in; (void)out_size; (void)ws_size;
  const float* x     = (const float*)d_in[0];
  const float* normw = (const float*)d_in[1];
  const float* wq    = (const float*)d_in[2];
  const float* wk    = (const float*)d_in[3];
  const float* wv    = (const float*)d_in[4];
  const float* k_pos = (const float*)d_in[5];
  const float* v_pos = (const float*)d_in[6];
  const float* memkv = (const float*)d_in[7];
  const float* k_w1  = (const float*)d_in[8];
  const float* k_b1  = (const float*)d_in[9];
  const float* k_w2  = (const float*)d_in[10];
  const float* k_b2  = (const float*)d_in[11];
  const float* v_w1  = (const float*)d_in[12];
  const float* v_b1  = (const float*)d_in[13];
  const float* v_w2  = (const float*)d_in[14];
  const float* v_b2  = (const float*)d_in[15];
  const float* g_w   = (const float*)d_in[16];
  const float* g_b   = (const float*)d_in[17];
  const float* wo    = (const float*)d_in[18];

  char* ws = (char*)d_ws; size_t off = 0;
  auto alloc = [&](size_t bytes) -> void* {
    void* p = ws + off; off = (off + bytes + 255) & ~(size_t)255; return p;
  };
  float*    h32   = (float*)alloc((size_t)S_ * D_ * 4);
  _Float16* h16   = (_Float16*)alloc((size_t)S_ * D_ * 2);
  _Float16* wqT   = (_Float16*)alloc((size_t)D_ * 1024 * 2);
  _Float16* wkT   = (_Float16*)alloc((size_t)D_ * 256 * 2);
  _Float16* wvT   = (_Float16*)alloc((size_t)D_ * 256 * 2);
  _Float16* woT   = (_Float16*)alloc((size_t)D_ * 1024 * 2);
  _Float16* gwT   = (_Float16*)alloc((size_t)D_ * 48 * 2);
  _Float16* kw1T  = (_Float16*)alloc((size_t)CHID_ * 2048 * 2);
  _Float16* kw2T  = (_Float16*)alloc((size_t)CHID_ * 64 * 2);
  _Float16* vw1T  = (_Float16*)alloc((size_t)CHID_ * 2048 * 2);
  _Float16* vw2T  = (_Float16*)alloc((size_t)CHID_ * 64 * 2);
  float*    q32   = (float*)alloc((size_t)S_ * 1024 * 4);
  float*    k32   = (float*)alloc((size_t)S_ * 256 * 4);
  float*    v32   = (float*)alloc((size_t)S_ * 256 * 4);
  _Float16* q16   = (_Float16*)alloc((size_t)S_ * 1024 * 2);
  _Float16* qr16  = (_Float16*)alloc((size_t)S_ * 1024 * 2);
  _Float16* kr16  = (_Float16*)alloc((size_t)S_ * 256 * 2);
  _Float16* vv16  = (_Float16*)alloc((size_t)S_ * 256 * 2);
  _Float16* kflat = (_Float16*)alloc((size_t)256 * 2048 * 2);
  _Float16* vflat = (_Float16*)alloc((size_t)256 * 2048 * 2);
  _Float16* khid  = (_Float16*)alloc((size_t)256 * 2048 * 2);
  _Float16* vhid  = (_Float16*)alloc((size_t)256 * 2048 * 2);
  float*    ckblk = (float*)alloc((size_t)256 * 64 * 4);
  float*    cvblk = (float*)alloc((size_t)256 * 64 * 4);
  _Float16* ck16  = (_Float16*)alloc((size_t)HKV_ * CKP_ * 64 * 2);
  _Float16* cvT16 = (_Float16*)alloc((size_t)HKV_ * 64 * CKP_ * 2);
  float*    imp   = (float*)alloc((size_t)HKV_ * S_ * W_ * 4);
  int*      idxal = (int*)alloc((size_t)HKV_ * S_ * (NSEL_ + 1) * 4);
  int*      dupm  = (int*)alloc((size_t)HKV_ * S_ * 4);
  float*    coutb = (float*)alloc((size_t)S_ * 1024 * 4);
  float*    foutb = (float*)alloc((size_t)S_ * 1024 * 4);
  float*    soutb = (float*)alloc((size_t)S_ * 1024 * 4);
  float*    glin  = (float*)alloc((size_t)S_ * 48 * 4);
  _Float16* ocat  = (_Float16*)alloc((size_t)S_ * 1024 * 2);

  // 1) rmsnorm
  k_rmsnorm<<<S_, 256, 0, stream>>>(x, normw, h32, h16);

  // 2) weight transpose + f16 convert
  auto tr = [&](const float* src, _Float16* dst, int K, int N) {
    int n = K * N;
    k_transpose_cvt<<<(n + 255) / 256, 256, 0, stream>>>(src, dst, K, N);
  };
  tr(wq, wqT, 1024, 1024);  tr(wk, wkT, 1024, 256);  tr(wv, wvT, 1024, 256);
  tr(wo, woT, 1024, 1024);  tr(g_w, gwT, 1024, 48);
  tr(k_w1, kw1T, 2048, 2048); tr(k_w2, kw2T, 2048, 64);
  tr(v_w1, vw1T, 2048, 2048); tr(v_w2, vw2T, 2048, 64);

  // 3) q/k/v + gate projections  (grid: ceil(N/64) x M/128, 128 threads)
  k_gemm<<<dim3(1024 / 64, S_ / 128), 128, 0, stream>>>(h16, 1024, wqT, 1024,
      q32, nullptr, 1024, nullptr, 0, S_, 1024, 1024);
  k_gemm<<<dim3(256 / 64, S_ / 128), 128, 0, stream>>>(h16, 1024, wkT, 1024,
      k32, nullptr, 256, nullptr, 0, S_, 256, 1024);
  k_gemm<<<dim3(256 / 64, S_ / 128), 128, 0, stream>>>(h16, 1024, wvT, 1024,
      v32, nullptr, 256, nullptr, 0, S_, 256, 1024);
  k_gemm<<<dim3(1, S_ / 128), 128, 0, stream>>>(h16, 1024, gwT, 1024,
      glin, nullptr, 48, g_b, 0, S_, 48, 1024);

  // 4) rotary + f16 packs
  k_rope_pack<<<S_, 512, 0, stream>>>(q32, k32, v32, q16, qr16, kr16, vv16);

  // 5) compress MLP (k and v)
  {
    int n = HKV_ * W_ * BS_ * DH_;
    k_compress_in<<<(n + 255) / 256, 256, 0, stream>>>(k32, k_pos, kflat);
    k_compress_in<<<(n + 255) / 256, 256, 0, stream>>>(v32, v_pos, vflat);
  }
  k_gemm<<<dim3(2048 / 64, 2), 128, 0, stream>>>(kflat, 2048, kw1T, 2048,
      nullptr, khid, 2048, k_b1, 1, 256, 2048, 2048);
  k_gemm<<<dim3(2048 / 64, 2), 128, 0, stream>>>(vflat, 2048, vw1T, 2048,
      nullptr, vhid, 2048, v_b1, 1, 256, 2048, 2048);
  k_gemm<<<dim3(1, 2), 128, 0, stream>>>(khid, 2048, kw2T, 2048,
      ckblk, nullptr, 64, k_b2, 0, 256, 64, 2048);
  k_gemm<<<dim3(1, 2), 128, 0, stream>>>(vhid, 2048, vw2T, 2048,
      cvblk, nullptr, 64, v_b2, 0, 256, 64, 2048);
  {
    int n = HKV_ * CKP_ * 64;
    k_pack_c<<<(n + 255) / 256, 256, 0, stream>>>(ckblk, memkv, 0, ck16, nullptr);
    k_pack_c<<<(n + 255) / 256, 256, 0, stream>>>(cvblk, memkv, 1, nullptr, cvT16);
  }

  // 6) compressed attention (+ block importance accumulation)
  {
    int n = HKV_ * S_ * W_;
    k_zero<<<(n + 255) / 256, 256, 0, stream>>>(imp, n);
  }
  k_compressed_attn<<<dim3(S_ / 16, G_, HKV_), 32, 0, stream>>>(q16, ck16, cvT16, imp, coutb);

  // 7) top-k block selection
  k_topk<<<dim3(S_ / 256, HKV_), 256, 0, stream>>>(imp, idxal, dupm);

  // 8) fine attention over selected blocks
  k_fine_attn<<<dim3(S_, HKV_), 32, 0, stream>>>(qr16, kr16, vv16, idxal, dupm, foutb);

  // 9) sliding-window attention
  k_window_attn<<<dim3(S_ / 16, G_, HKV_), 32, 0, stream>>>(qr16, kr16, vv16, soutb);

  // 10) gated combine + output projection
  {
    int n = S_ * D_;
    k_combine<<<(n + 255) / 256, 256, 0, stream>>>(coutb, foutb, soutb, glin, ocat);
  }
  k_gemm<<<dim3(1024 / 64, S_ / 128), 128, 0, stream>>>(ocat, 1024, woT, 1024,
      (float*)d_out, nullptr, 1024, nullptr, 0, S_, 1024, 1024);
}
